// MoEMiniGPT_12335146074167
// MI455X (gfx1250) — matrix operand order repaired and checked by reading, not compiled, not run on hardware
//
#include <hip/hip_runtime.h>
#include <hip/hip_bf16.h>
#include <math.h>

// ---------------------------------------------------------------------------
// MoE MiniGPT layer for MI455X (gfx1250, wave32, WMMA).
// Sparse top-2 formulation (mathematically identical to the dense reference).
// Pipeline:
//   0) zero output / ws accumulators
//   1) transpose+convert expert weights f32 -> f16 in ws (one-time, HBM-bound)
//   2) router (wave32 per token) -> per-expert token lists + loss partials
//   3) scalar losses
//   4) grouped per-expert FFN with v_wmma_f32_16x16x32_f16:
//        B frags: 2x global_load_b128 from pre-transposed f16 weights
//        A frags: 2x ds_load_b128 from LDS, hoisted across N-tiles
// ---------------------------------------------------------------------------

#define NB 4
#define NS 1024
#define NTOK (NB * NS)      // 4096 tokens
#define DD 768              // model dim
#define FF 3072             // ffn dim
#define NE 8                // experts
#define FC 512              // F-chunk staged in LDS per iteration
#define BSD (NTOK * DD)

typedef __attribute__((ext_vector_type(16))) _Float16 v16h;
typedef __attribute__((ext_vector_type(8)))  _Float16 v8h;
typedef __attribute__((ext_vector_type(8)))  float    v8f;

// ---------------------------------------------------------------------------
// Kernel 0: zero output accumulation region + workspace accumulators
// ---------------------------------------------------------------------------
__global__ void moe_init_kernel(float* __restrict__ out, int n,
                                int* __restrict__ counts,
                                float* __restrict__ usage,
                                float* __restrict__ zsum) {
  int i = blockIdx.x * blockDim.x + threadIdx.x;
  if (i < n) out[i] = 0.0f;
  if (blockIdx.x == 0) {
    if (threadIdx.x < NE) { counts[threadIdx.x] = 0; usage[threadIdx.x] = 0.0f; }
    if (threadIdx.x == NE) zsum[0] = 0.0f;
  }
}

// ---------------------------------------------------------------------------
// Kernel 1: tiled transpose + f32->f16 convert:  src[R][C] -> dst[C][R]
// (per-expert slab selected by blockIdx.z). Coalesced on both sides via LDS.
// ---------------------------------------------------------------------------
__global__ __launch_bounds__(256) void moe_tcvt_kernel(
    const float* __restrict__ src, _Float16* __restrict__ dst, int R, int C) {
  __shared__ _Float16 t[32][33];
  const size_t slab = (size_t)blockIdx.z * R * C;
  src += slab;
  dst += slab;
  const int c0 = blockIdx.x * 32;
  const int r0 = blockIdx.y * 32;
#pragma unroll
  for (int i = threadIdx.y; i < 32; i += 8)
    t[i][threadIdx.x] = (_Float16)src[(size_t)(r0 + i) * C + c0 + threadIdx.x];
  __syncthreads();
#pragma unroll
  for (int i = threadIdx.y; i < 32; i += 8)
    dst[(size_t)(c0 + i) * R + r0 + threadIdx.x] = t[threadIdx.x][i];
}

// ---------------------------------------------------------------------------
// Kernel 2: router. One wave32 per token.
// ---------------------------------------------------------------------------
__global__ __launch_bounds__(128) void moe_router_kernel(
    const float* __restrict__ x, const float* __restrict__ Wr,
    int* __restrict__ counts, int* __restrict__ listIdx,
    float* __restrict__ listW, float* __restrict__ usage,
    float* __restrict__ zsum) {
  const int wave = threadIdx.x >> 5;
  const int lane = threadIdx.x & 31;
  const int t = blockIdx.x * (blockDim.x >> 5) + wave;
  if (t >= NTOK) return;

  float acc[NE];
#pragma unroll
  for (int e = 0; e < NE; ++e) acc[e] = 0.0f;

  const float* xr = x + (size_t)t * DD;
  for (int d = lane; d < DD; d += 32) {
    float xv = xr[d];
    const float* wr = Wr + (size_t)d * NE;
#pragma unroll
    for (int e = 0; e < NE; ++e) acc[e] += xv * wr[e];
  }
#pragma unroll
  for (int e = 0; e < NE; ++e) {
#pragma unroll
    for (int off = 16; off >= 1; off >>= 1) acc[e] += __shfl_xor(acc[e], off, 32);
  }

  if (lane == 0) {
    float m = acc[0];
#pragma unroll
    for (int e = 1; e < NE; ++e) m = fmaxf(m, acc[e]);
    float p[NE], s = 0.0f;
#pragma unroll
    for (int e = 0; e < NE; ++e) { p[e] = __expf(acc[e] - m); s += p[e]; }
    float inv = 1.0f / s;
#pragma unroll
    for (int e = 0; e < NE; ++e) p[e] *= inv;
    float lse = m + __logf(s);

    // top-2 with first-index-wins ties (matches jax.lax.top_k)
    int e0 = 0;
#pragma unroll
    for (int e = 1; e < NE; ++e) if (p[e] > p[e0]) e0 = e;
    int e1 = (e0 == 0) ? 1 : 0;
#pragma unroll
    for (int e = 0; e < NE; ++e) if (e != e0 && p[e] > p[e1]) e1 = e;

    float denom = p[e0] + p[e1] + 1e-8f;
    float w0 = p[e0] / denom;
    float w1c = p[e1] / denom;

#pragma unroll
    for (int e = 0; e < NE; ++e) atomicAdd(&usage[e], p[e]);
    atomicAdd(zsum, lse * lse);

    int pos0 = atomicAdd(&counts[e0], 1);
    listIdx[e0 * NTOK + pos0] = t;
    listW[e0 * NTOK + pos0] = w0;
    int pos1 = atomicAdd(&counts[e1], 1);
    listIdx[e1 * NTOK + pos1] = t;
    listW[e1 * NTOK + pos1] = w1c;
  }
}

// ---------------------------------------------------------------------------
// Kernel 3: scalar auxiliary losses
// ---------------------------------------------------------------------------
__global__ void moe_loss_kernel(const float* __restrict__ usage,
                                const float* __restrict__ zsum,
                                float* __restrict__ out_losses) {
  if (threadIdx.x == 0 && blockIdx.x == 0) {
    const float invN = 1.0f / (float)NTOK;
    float u[NE], mean = 0.0f;
#pragma unroll
    for (int e = 0; e < NE; ++e) { u[e] = usage[e] * invN; mean += u[e]; }
    mean *= (1.0f / NE);
    float var = 0.0f;
#pragma unroll
    for (int e = 0; e < NE; ++e) { float d = u[e] - mean; var += d * d; }
    var *= (1.0f / NE);
    out_losses[0] = var / (mean * mean + 1e-8f) * (float)NE * 0.01f;
    out_losses[1] = zsum[0] * invN * 0.001f;
  }
}

// ---------------------------------------------------------------------------
// FFN fragment helpers
// ---------------------------------------------------------------------------
__device__ inline v16h load_a_frag(const _Float16* __restrict__ row, int k0, int half) {
  // A fragment (16x32 f16): lane holds its M-row, K = {k0+h*16+0..15}
  v8h lo = *(const v8h*)(row + k0 + half * 16);
  v8h hi = *(const v8h*)(row + k0 + half * 16 + 8);
  return __builtin_shufflevector(lo, hi, 0, 1, 2, 3, 4, 5, 6, 7, 8, 9, 10, 11, 12, 13, 14, 15);
}

__device__ inline v16h load_b_frag(const _Float16* __restrict__ p) {
  // B fragment (32x16 f16): lane holds its N-column, 16 consecutive K starting
  // at (lane>>4)*16 -- contiguous in the pre-transposed f16 weight layout.
  v8h lo = *(const v8h*)p;
  v8h hi = *(const v8h*)(p + 8);
  return __builtin_shufflevector(lo, hi, 0, 1, 2, 3, 4, 5, 6, 7, 8, 9, 10, 11, 12, 13, 14, 15);
}

// Branch-free erf, Abramowitz-Stegun 7.1.26 (|err| <= 1.5e-7). Uses only
// v_rcp/v_exp trans ops: no EXEC divergence between WMMAs.
__device__ inline float erf_fast(float v) {
  float a = fabsf(v);
  float t = 1.0f / (1.0f + 0.3275911f * a);
  float y = ((((1.061405429f * t - 1.453152027f) * t) + 1.421413741f) * t -
             0.284496736f) * t + 0.254829592f;
  y = 1.0f - y * t * __expf(-a * a);
  return copysignf(y, v);
}

// ---------------------------------------------------------------------------
// Kernel 4: grouped per-expert FFN.
// Block = (expert e, 16-token tile). 8 waves, wave32.
//   gemm1: h = gelu(x_tile @ w1[e] + b1[e]) in F-chunks of 512 -> LDS f16
//   gemm2: acc += h_chunk @ w2[e]   (persistent v8f accumulators per wave)
//   epilogue: out[token] += cw * (acc + b2[e])
// A fragments hoisted across N-tiles (k-outer, j-inner).
// ---------------------------------------------------------------------------
__global__ __launch_bounds__(256) void moe_ffn_kernel(
    const float* __restrict__ x,
    const _Float16* __restrict__ w1t,   // [E][FF][DD] f16 (transposed)
    const float* __restrict__ b1,
    const _Float16* __restrict__ w2t,   // [E][DD][FF] f16 (transposed)
    const float* __restrict__ b2, const int* __restrict__ counts,
    const int* __restrict__ listIdx, const float* __restrict__ listW,
    float* __restrict__ out) {
  const int e = blockIdx.y;
  const int cnt = counts[e];
  const int base = blockIdx.x * 16;
  if (base >= cnt) return;

  __shared__ __align__(16) _Float16 xs[16][DD + 8];   // x tile, f16, padded rows
  __shared__ __align__(16) _Float16 hs[16][FC + 8];   // gelu(h) chunk, f16
  __shared__ int stok[16];
  __shared__ float scw[16];

  const int tid = threadIdx.x;
  const int wave = tid >> 5;
  const int lane = tid & 31;
  const int half = lane >> 4;
  const int nl = lane & 15;

  if (tid < 16) {
    int idx = base + tid;
    if (idx < cnt) {
      stok[tid] = listIdx[e * NTOK + idx];
      scw[tid] = listW[e * NTOK + idx];
    } else {
      stok[tid] = -1;
      scw[tid] = 0.0f;
    }
  }
  __syncthreads();

  for (int i = tid; i < 16 * DD; i += 256) {
    int r = i / DD, d = i - r * DD;
    int t = stok[r];
    xs[r][d] = (t >= 0) ? (_Float16)x[(size_t)t * DD + d] : (_Float16)0.0f;
  }
  __syncthreads();

  const _Float16* W1T = w1t + (size_t)e * FF * DD;
  const _Float16* W2T = w2t + (size_t)e * DD * FF;
  const float* B1 = b1 + e * FF;
  const float* B2 = b2 + e * DD;

  const v8f vzero = {0.0f, 0.0f, 0.0f, 0.0f, 0.0f, 0.0f, 0.0f, 0.0f};

  v8f acc2[6];
#pragma unroll
  for (int j = 0; j < 6; ++j) acc2[j] = vzero;

  for (int fc = 0; fc < FF; fc += FC) {
    // ---- gemm1 chunk: 32 n-tiles over 8 waves (4 each), k-outer ----
    v8f acc1[4];
#pragma unroll
    for (int j = 0; j < 4; ++j) acc1[j] = vzero;

    const int n1base = fc + wave * 64 + nl;  // this wave's first F column
    for (int k0 = 0; k0 < DD; k0 += 32) {
      v16h a = load_a_frag(&xs[nl][0], k0, half);
      const int kb = k0 + half * 16;
#pragma unroll
      for (int j = 0; j < 4; ++j) {
        const _Float16* Bp = W1T + (size_t)(n1base + j * 16) * DD + kb;
        acc1[j] = __builtin_amdgcn_wmma_f32_16x16x32_f16(
            false, a, false, load_b_frag(Bp), (short)0, acc1[j], false, false);
      }
    }
    // bias + exact GELU -> LDS f16
#pragma unroll
    for (int j = 0; j < 4; ++j) {
      const int nloc = wave * 64 + j * 16 + nl;  // column within chunk [0,512)
      const float bn = B1[fc + nloc];
#pragma unroll
      for (int q = 0; q < 8; ++q) {
        float v = acc1[j][q] + bn;
        float g = 0.5f * v * (1.0f + erf_fast(v * 0.70710678118654752f));
        hs[q + 8 * half][nloc] = (_Float16)g;
      }
    }
    __syncthreads();

    // ---- gemm2 partial: 48 n-tiles over 8 waves (6 each), k-outer ----
    for (int k0 = 0; k0 < FC; k0 += 32) {
      v16h a = load_a_frag(&hs[nl][0], k0, half);
      const int kb = fc + k0 + half * 16;
#pragma unroll
      for (int j = 0; j < 6; ++j) {
        const int n = (wave * 6 + j) * 16 + nl;  // global D column
        const _Float16* Bp = W2T + (size_t)n * FF + kb;
        acc2[j] = __builtin_amdgcn_wmma_f32_16x16x32_f16(
            false, a, false, load_b_frag(Bp), (short)0, acc2[j], false, false);
      }
    }
    __syncthreads();
  }

  // ---- epilogue: bias, combine weight, atomic scatter ----
#pragma unroll
  for (int j = 0; j < 6; ++j) {
    const int n = (wave * 6 + j) * 16 + nl;
    const float bn = B2[n];
#pragma unroll
    for (int q = 0; q < 8; ++q) {
      const int m = q + 8 * half;
      const int t = stok[m];
      if (t >= 0) atomicAdd(&out[(size_t)t * DD + n], scw[m] * (acc2[j][q] + bn));
    }
  }
}

// ---------------------------------------------------------------------------
// Host launch
// ---------------------------------------------------------------------------
extern "C" void kernel_launch(void* const* d_in, const int* in_sizes, int n_in,
                              void* d_out, int out_size, void* d_ws, size_t ws_size,
                              hipStream_t stream) {
  const float* x  = (const float*)d_in[0];
  const float* Wr = (const float*)d_in[1];
  const float* w1 = (const float*)d_in[2];
  const float* b1 = (const float*)d_in[3];
  const float* w2 = (const float*)d_in[4];
  const float* b2 = (const float*)d_in[5];
  float* out = (float*)d_out;

  char* ws = (char*)d_ws;
  int*   counts  = (int*)ws;                              // 8 ints
  float* usage   = (float*)(ws + 32);                     // 8 floats
  float* zsum    = (float*)(ws + 64);                     // 1 float
  int*   listIdx = (int*)(ws + 128);                      // 8*4096 ints
  float* listW   = (float*)(ws + 128 + NE * NTOK * 4);    // 8*4096 floats
  // f16 transposed weights (37.75 MB each), 1 MB aligned region
  _Float16* w1t = (_Float16*)(ws + (1ull << 20));
  _Float16* w2t = w1t + (size_t)NE * FF * DD;

  moe_init_kernel<<<dim3((BSD + 255) / 256), 256, 0, stream>>>(out, BSD, counts, usage, zsum);
  // w1 [E][DD][FF] -> w1t [E][FF][DD]
  moe_tcvt_kernel<<<dim3(FF / 32, DD / 32, NE), dim3(32, 8), 0, stream>>>(w1, w1t, DD, FF);
  // w2 [E][FF][DD] -> w2t [E][DD][FF]
  moe_tcvt_kernel<<<dim3(DD / 32, FF / 32, NE), dim3(32, 8), 0, stream>>>(w2, w2t, FF, DD);
  moe_router_kernel<<<dim3(NTOK / 4), 128, 0, stream>>>(x, Wr, counts, listIdx, listW, usage, zsum);
  moe_loss_kernel<<<1, 32, 0, stream>>>(usage, zsum, out + BSD);
  moe_ffn_kernel<<<dim3(NTOK / 16, NE), 256, 0, stream>>>(x, w1t, b1, w2t, b2, counts,
                                                          listIdx, listW, out);
}
